// MSHGTModel_54331336295106
// MI455X (gfx1250) — compile-verified, matrix operand order checked by reading, not run on hardware
//
#include <hip/hip_runtime.h>
#include <math.h>

// ---------------------------------------------------------------------------
// CDNA5 (gfx1250) implementation of the MSHGT model.
// All dense matmuls run through v_wmma_f32_16x16x32_bf16 (wave32 WMMA),
// attention is flash-style (online softmax, no materialized 4096x4096 scores).
// ---------------------------------------------------------------------------

typedef __attribute__((ext_vector_type(16))) __bf16 v16bf;
typedef __attribute__((ext_vector_type(8)))  float  v8f;
typedef __attribute__((ext_vector_type(8)))  unsigned short vus8;

#define WMMA_BF16 __builtin_amdgcn_wmma_f32_16x16x32_bf16

__device__ __forceinline__ unsigned short f2bf(float f) {
  unsigned u = __float_as_uint(f);
  u += 0x7FFFu + ((u >> 16) & 1u);        // round-to-nearest-even
  return (unsigned short)(u >> 16);
}

// A-operand K mapping (ISA 7.12.2, 16-bit A 16x32): lanes 0-15 hold K 0-7,16-23;
// lanes 16-31 hold K 8-15,24-31. element e -> K.
__device__ __forceinline__ int kmapE(int e, int half) {
  return ((e < 8) ? e : e + 8) + half * 8;
}

__device__ __forceinline__ float atomicMaxF(float* addr, float val) {
  if (val >= 0.f)
    return __int_as_float(atomicMax((int*)addr, __float_as_int(val)));
  return __uint_as_float(atomicMin((unsigned int*)addr, __float_as_uint(val)));
}

// ---------------------------------------------------------------------------
// Generic WMMA GEMM: C[M,N] = act(scale * A[M,K] @ B + bias)
// B is [K,N] (transB=0) or [N,K] (transB=1, used for pe_Q @ pe_K^T).
// Block = 128 threads = 4 waves; block tile = 32 rows x 64 cols
// (2 row-tiles per wave sharing one B fragment -> 2 WMMAs per K-step).
// A/B staged in LDS pre-swizzled into WMMA fragment order so every lane
// reads one contiguous 32B v16bf and every stager writes contiguous 16B.
// Requires M%32==0 and K%32==0 (true for every GEMM in this model).
// ---------------------------------------------------------------------------
__global__ __launch_bounds__(128)
void gemm_wmma(const float* __restrict__ A, const float* __restrict__ B,
               const float* __restrict__ bias, float* __restrict__ C,
               int M, int N, int Kdim, int transB, int act, float scale)
{
  __shared__ __align__(32) unsigned short sA[2 * 16 * 32];  // 2 row-tiles
  __shared__ __align__(32) unsigned short sB[4 * 32 * 16];  // 4 col-groups
  const int t    = threadIdx.x;
  const int lane = t & 31;
  const int wave = t >> 5;
  const int m0 = blockIdx.y * 32;
  const int n0 = blockIdx.x * 64;

  // per-thread staging coordinates (constant over the K loop)
  const int ar  = t >> 2;              // A row 0..31
  const int ac0 = (t & 3) * 8;         // A col group (8 contiguous K)
  const int ahl = (ac0 >> 3) & 1;
  const int aeb = (ac0 >= 16) ? 8 : 0;
  const int apos = (((ar >> 4) * 32 + ahl * 16 + (ar & 15)) * 16) + aeb;

  const int bcol = n0 + (t & 63);      // B column
  const int bhl  = t >> 6;             // lane-half (K split)
  const int bpos = (((t & 63) >> 4) * 32 + bhl * 16 + (t & 15)) * 16;

  v8f acc0 = {}, acc1 = {};
  for (int k0 = 0; k0 < Kdim; k0 += 32) {
    if (k0 + 32 < Kdim)   // speculative L2 prefetch of next A tile
      __builtin_prefetch(A + (size_t)(m0 + ar) * Kdim + k0 + 32 + ac0, 0, 1);
    __syncthreads();
    // ---- stage A tile (32 rows x 32 K): 2x float4 load -> 1x 16B LDS store
    {
      const float* ap = A + (size_t)(m0 + ar) * Kdim + k0 + ac0;
      float4 f0 = *(const float4*)(ap);
      float4 f1 = *(const float4*)(ap + 4);
      union { vus8 v; unsigned short u[8]; } pk;
      pk.u[0] = f2bf(f0.x); pk.u[1] = f2bf(f0.y);
      pk.u[2] = f2bf(f0.z); pk.u[3] = f2bf(f0.w);
      pk.u[4] = f2bf(f1.x); pk.u[5] = f2bf(f1.y);
      pk.u[6] = f2bf(f1.z); pk.u[7] = f2bf(f1.w);
      *(vus8*)&sA[apos] = pk.v;
    }
    // ---- stage B tile (32 K x 64 cols): thread owns (col, K-half)
    {
      unsigned short u[16];
      if (bcol < N) {
        if (transB) {
          const float* bp = B + (size_t)bcol * Kdim + k0 + bhl * 8;
          float4 a0 = *(const float4*)(bp);
          float4 a1 = *(const float4*)(bp + 4);
          float4 a2 = *(const float4*)(bp + 16);
          float4 a3 = *(const float4*)(bp + 20);
          u[0] = f2bf(a0.x);  u[1] = f2bf(a0.y);  u[2]  = f2bf(a0.z);  u[3]  = f2bf(a0.w);
          u[4] = f2bf(a1.x);  u[5] = f2bf(a1.y);  u[6]  = f2bf(a1.z);  u[7]  = f2bf(a1.w);
          u[8] = f2bf(a2.x);  u[9] = f2bf(a2.y);  u[10] = f2bf(a2.z);  u[11] = f2bf(a2.w);
          u[12] = f2bf(a3.x); u[13] = f2bf(a3.y); u[14] = f2bf(a3.z);  u[15] = f2bf(a3.w);
        } else {
#pragma unroll
          for (int e = 0; e < 16; ++e)
            u[e] = f2bf(B[(size_t)(k0 + kmapE(e, bhl)) * N + bcol]);
        }
      } else {
#pragma unroll
        for (int e = 0; e < 16; ++e) u[e] = 0;
      }
      union { vus8 v; unsigned short s[8]; } p0, p1;
#pragma unroll
      for (int e = 0; e < 8; ++e) { p0.s[e] = u[e]; p1.s[e] = u[e + 8]; }
      *(vus8*)&sB[bpos]     = p0.v;
      *(vus8*)&sB[bpos + 8] = p1.v;
    }
    __syncthreads();
    v16bf af0 = *(const v16bf*)&sA[lane * 16];
    v16bf af1 = *(const v16bf*)&sA[(32 + lane) * 16];
    v16bf bf  = *(const v16bf*)&sB[(wave * 32 + lane) * 16];
    acc0 = WMMA_BF16(false, af0, false, bf, (short)0, acc0, false, false);
    acc1 = WMMA_BF16(false, af1, false, bf, (short)0, acc1, false, false);
  }

  // C layout (ISA 7.12.2): lane n = N col, VGPR v -> row v (+8 for upper half)
  const int n    = lane & 15;
  const int half = lane >> 4;
  const int col  = n0 + wave * 16 + n;
  if (col < N) {
    float b = bias ? bias[col] : 0.f;
#pragma unroll
    for (int r = 0; r < 8; ++r) {
      int row = m0 + r + half * 8;
      float x0 = acc0[r] * scale + b;
      float x1 = acc1[r] * scale + b;
      if (act == 1) { x0 = fmaxf(x0, 0.f); x1 = fmaxf(x1, 0.f); }
      C[(size_t)row * N + col]        = x0;
      C[(size_t)(row + 16) * N + col] = x1;
    }
  }
}

// ---------------------------------------------------------------------------
// Flash attention for one head: O = softmax(Q K^T / sqrt(32) + pe/sqrt(32)
//                                           + degb) V
// Block = 256 threads = 8 waves; wave owns a 16-row tile; K/V chunks (32 cols)
// staged cooperatively in LDS in B-fragment order. Online softmax state
// (row max / row sum) kept in registers, 16-lane butterfly reductions.
// ---------------------------------------------------------------------------
__global__ __launch_bounds__(256)
void flash_attn(const float* __restrict__ Q, const float* __restrict__ Km,
                const float* __restrict__ Vm, const float* __restrict__ pe,
                const float* __restrict__ degb, float* __restrict__ Out,
                int N, float sscale, float pescale)
{
  __shared__ __align__(32) unsigned short sKT[2 * 32 * 16]; // K^T frags, 2 col-groups
  __shared__ __align__(32) unsigned short sV [2 * 32 * 16]; // V frags, 2 dh-halves
  __shared__ __align__(32) unsigned short sP [8 * 32 * 16]; // P A-frags, per wave
  const int t = threadIdx.x, lane = t & 31, wave = t >> 5;
  const int head = blockIdx.y;
  const int r0   = blockIdx.x * 128 + wave * 16;
  const int hoff = head * 32;
  const int n    = lane & 15;
  const int half = lane >> 4;

  // Q A-fragment for this wave's 16 rows (dh = 32 -> single K step)
  union { v16bf v; unsigned short u[16]; } qf;
  {
    const float* qp = Q + (size_t)(r0 + n) * 128 + hoff;
#pragma unroll
    for (int e = 0; e < 16; ++e) qf.u[e] = f2bf(qp[kmapE(e, half)]);
  }

  v8f O0 = {}, O1 = {};
  float mrow[8], lrow[8];
#pragma unroll
  for (int i = 0; i < 8; ++i) { mrow[i] = -3.0e38f; lrow[i] = 0.f; }

  for (int c0 = 0; c0 < N; c0 += 32) {
    if (c0 + 32 < N)       // prefetch next pe-bias chunk (dominant traffic)
      __builtin_prefetch(pe + (size_t)(r0 + half * 8) * N + c0 + 32 + n, 0, 1);
    __syncthreads();
    // stage K^T (cols c0..c0+31 x dh32) and V (tokens c0..c0+31 x dh32)
#pragma unroll
    for (int j = 0; j < 4; ++j) {
      int i  = t * 4 + j;
      int g  = i >> 9, rem = i & 511;
      int ln = rem >> 4, e = rem & 15;
      int nn = ln & 15, hh = ln >> 4;
      int k  = kmapE(e, hh);
      sKT[i] = f2bf(Km[(size_t)(c0 + g * 16 + nn) * 128 + hoff + k]);
      sV [i] = f2bf(Vm[(size_t)(c0 + k) * 128 + hoff + g * 16 + nn]);
    }
    __syncthreads();

    v16bf kt0 = *(const v16bf*)&sKT[0 * 512 + lane * 16];
    v16bf kt1 = *(const v16bf*)&sKT[1 * 512 + lane * 16];
    v8f S0 = {}, S1 = {};
    S0 = WMMA_BF16(false, qf.v, false, kt0, (short)0, S0, false, false);
    S1 = WMMA_BF16(false, qf.v, false, kt1, (short)0, S1, false, false);

    float db0 = degb[c0 + n];
    float db1 = degb[c0 + 16 + n];
#pragma unroll
    for (int r = 0; r < 8; ++r) {
      const float* per = pe + (size_t)(r0 + r + half * 8) * N;
      S0[r] = S0[r] * sscale + per[c0 + n]      * pescale + db0;
      S1[r] = S1[r] * sscale + per[c0 + 16 + n] * pescale + db1;
    }
    // online softmax update per row (reductions stay inside each 16-lane half)
    unsigned short* pw = &sP[wave * 512];
#pragma unroll
    for (int r = 0; r < 8; ++r) {
      float tmax = fmaxf(S0[r], S1[r]);
#pragma unroll
      for (int off = 1; off < 16; off <<= 1) tmax = fmaxf(tmax, __shfl_xor(tmax, off, 32));
      float mnew  = fmaxf(mrow[r], tmax);
      float alpha = __expf(mrow[r] - mnew);
      mrow[r] = mnew;
      lrow[r] *= alpha;
      O0[r] *= alpha; O1[r] *= alpha;
      S0[r] = __expf(S0[r] - mnew);
      S1[r] = __expf(S1[r] - mnew);
      float ts = S0[r] + S1[r];
#pragma unroll
      for (int off = 1; off < 16; off <<= 1) ts += __shfl_xor(ts, off, 32);
      lrow[r] += ts;
      // scatter P into A-fragment order (row rr, col c)
      int rr = r + half * 8;
      { int c = n;      int hl = (c >> 3) & 1; int e = (c & 7) + ((c >= 16) ? 8 : 0);
        pw[(hl * 16 + rr) * 16 + e] = f2bf(S0[r]); }
      { int c = 16 + n; int hl = (c >> 3) & 1; int e = (c & 7) + ((c >= 16) ? 8 : 0);
        pw[(hl * 16 + rr) * 16 + e] = f2bf(S1[r]); }
    }
    __syncthreads();
    v16bf pf = *(const v16bf*)&pw[lane * 16];
    v16bf v0 = *(const v16bf*)&sV[0 * 512 + lane * 16];
    v16bf v1 = *(const v16bf*)&sV[1 * 512 + lane * 16];
    O0 = WMMA_BF16(false, pf, false, v0, (short)0, O0, false, false);
    O1 = WMMA_BF16(false, pf, false, v1, (short)0, O1, false, false);
  }

#pragma unroll
  for (int r = 0; r < 8; ++r) {
    int row = r0 + r + half * 8;
    float inv = 1.0f / lrow[r];
    Out[(size_t)row * 128 + hoff + n]      = O0[r] * inv;
    Out[(size_t)row * 128 + hoff + 16 + n] = O1[r] * inv;
  }
}

// ---------------------------------------------------------------------------
// Small support kernels
// ---------------------------------------------------------------------------
__global__ void fill_kernel(float* p, float v, size_t n) {
  size_t i = (size_t)blockIdx.x * blockDim.x + threadIdx.x;
  if (i < n) p[i] = v;
}
__global__ void relu_kernel(float* p, size_t n) {
  size_t i = (size_t)blockIdx.x * blockDim.x + threadIdx.x;
  if (i < n) p[i] = fmaxf(p[i], 0.f);
}
__global__ void addbias_kernel(float* p, const float* __restrict__ b, int rows, int cols) {
  size_t i = (size_t)blockIdx.x * blockDim.x + threadIdx.x;
  if (i < (size_t)rows * cols) p[i] += b[i % cols];
}
// a_s[n,h] = sum_c h[n, h*C + c] * a[h, c]
__global__ void alpha_red(const float* __restrict__ h, const float* __restrict__ a,
                          float* __restrict__ out, int Nn, int H, int C) {
  int i = blockIdx.x * blockDim.x + threadIdx.x;
  if (i >= Nn * H) return;
  int node = i / H, hh = i % H;
  const float* hp = h + (size_t)node * H * C + hh * C;
  const float* ap = a + hh * C;
  float s = 0.f;
  for (int c = 0; c < C; ++c) s += hp[c] * ap[c];
  out[i] = s;
}
__global__ void edge_max(const int* __restrict__ src, const int* __restrict__ dst,
                         const float* __restrict__ as, const float* __restrict__ ad,
                         float* __restrict__ ebuf, float* __restrict__ mbuf, int E, int H) {
  int i = blockIdx.x * blockDim.x + threadIdx.x;
  if (i >= E * H) return;
  int e = i / H, h = i % H;
  float x = as[src[e] * H + h] + ad[dst[e] * H + h];
  x = (x >= 0.f) ? x : 0.2f * x;                    // leaky_relu(0.2)
  ebuf[i] = x;
  atomicMaxF(&mbuf[dst[e] * H + h], x);
}
__global__ void edge_expsum(const int* __restrict__ dst, const float* __restrict__ ebuf,
                            const float* __restrict__ mbuf, float* __restrict__ zbuf,
                            float* __restrict__ wbuf, int E, int H) {
  int i = blockIdx.x * blockDim.x + threadIdx.x;
  if (i >= E * H) return;
  int e = i / H, h = i % H;
  float w = __expf(ebuf[i] - mbuf[dst[e] * H + h]);
  wbuf[i] = w;
  atomicAdd(&zbuf[dst[e] * H + h], w);
}
__global__ void edge_agg(const int* __restrict__ src, const int* __restrict__ dst,
                         const float* __restrict__ wbuf, const float* __restrict__ zbuf,
                         const float* __restrict__ hsrc, float* __restrict__ out,
                         int E, int H, int C) {
  size_t i = (size_t)blockIdx.x * blockDim.x + threadIdx.x;
  if (i >= (size_t)E * H * C) return;
  int c = (int)(i % C); size_t eh = i / C;
  int h = (int)(eh % H); int e = (int)(eh / H);
  int s = src[e], d = dst[e];
  float w = wbuf[e * H + h] / (zbuf[d * H + h] + 1e-16f);
  atomicAdd(&out[(size_t)d * H * C + h * C + c], w * hsrc[(size_t)s * H * C + h * C + c]);
}
__global__ void ln_residual(const float* __restrict__ x, const float* __restrict__ r,
                            const float* __restrict__ g, const float* __restrict__ b,
                            float* __restrict__ y, int C) {
  __shared__ float s1[128], s2[128];
  int row = blockIdx.x, t = threadIdx.x;
  float v = x[(size_t)row * C + t] + r[(size_t)row * C + t];
  s1[t] = v; s2[t] = v * v;
  __syncthreads();
  for (int off = 64; off > 0; off >>= 1) {
    if (t < off) { s1[t] += s1[t + off]; s2[t] += s2[t + off]; }
    __syncthreads();
  }
  float mu = s1[0] / C;
  float var = s2[0] / C - mu * mu;
  float rs = rsqrtf(var + 1e-5f);
  y[(size_t)row * C + t] = (v - mu) * rs * g[t] + b[t];
}
__global__ void pe_loss(const float* __restrict__ Ahat, const float* __restrict__ Aorig,
                        float* __restrict__ acc, size_t NN) {
  size_t i = (size_t)blockIdx.x * blockDim.x + threadIdx.x;
  size_t stride = (size_t)gridDim.x * blockDim.x;
  float s = 0.f;
  for (size_t j = i; j < 4 * NN; j += stride) {
    float d = Ahat[j & (NN - 1)] - Aorig[j];
    s += d * d;
  }
  __shared__ float sh[256];
  sh[threadIdx.x] = s; __syncthreads();
  for (int off = 128; off > 0; off >>= 1) {
    if (threadIdx.x < off) sh[threadIdx.x] += sh[threadIdx.x + off];
    __syncthreads();
  }
  if (threadIdx.x == 0) atomicAdd(acc, sh[0]);
}
__global__ void finalize_loss(const float* acc, const float* alpha, float* out0, float invc) {
  out0[0] = alpha[0] * acc[0] * invc;
}
__global__ void log1p_kernel(const float* __restrict__ in, float* __restrict__ out, int n) {
  int i = blockIdx.x * blockDim.x + threadIdx.x;
  if (i < n) out[i] = log1pf(in[i]);
}
__global__ void concat3(const float* __restrict__ a, const float* __restrict__ b,
                        const float* __restrict__ c, float* __restrict__ cat, int rows) {
  size_t i = (size_t)blockIdx.x * blockDim.x + threadIdx.x;
  if (i >= (size_t)rows * 384) return;
  int n = (int)(i / 384), j = (int)(i % 384);
  int s = j / 128, ch = j % 128;
  const float* src = (s == 0) ? a : (s == 1) ? b : c;
  cat[i] = src[(size_t)n * 128 + ch];
}
__global__ void softmax5(const float* __restrict__ in, float* __restrict__ out, int rows) {
  int r = blockIdx.x * blockDim.x + threadIdx.x;
  if (r >= rows) return;
  float v[5]; float m = -3e38f;
  for (int c = 0; c < 5; ++c) { v[c] = in[r * 5 + c]; m = fmaxf(m, v[c]); }
  float s = 0.f;
  for (int c = 0; c < 5; ++c) { v[c] = __expf(v[c] - m); s += v[c]; }
  float inv = 1.f / s;
  for (int c = 0; c < 5; ++c) out[r * 5 + c] = v[c] * inv;
}

// ---------------------------------------------------------------------------
// Host orchestration
// ---------------------------------------------------------------------------
extern "C" void kernel_launch(void* const* d_in, const int* in_sizes, int n_in,
                              void* d_out, int out_size, void* d_ws, size_t ws_size,
                              hipStream_t stream) {
  (void)in_sizes; (void)n_in; (void)out_size; (void)ws_size;
  const int NM = 4096, ND = 1024, NA = 3072, IN = 512, HID = 128, H4C = 512;
  const size_t NN = (size_t)NM * NM;

  // ---- inputs (setup_inputs order, params pytree-flattened: sorted dict keys)
  const float* x_movie    = (const float*)d_in[0];
  const float* x_director = (const float*)d_in[1];
  const float* x_actor    = (const float*)d_in[2];
  const int*   ei[4]      = { (const int*)d_in[3], (const int*)d_in[4],
                              (const int*)d_in[5], (const int*)d_in[6] };
  const float* origA      = (const float*)d_in[7];
  const float* deg        = (const float*)d_in[8];
  const float* alpha_p    = (const float*)d_in[9];
  // blocks[b]: Wk Wo Wq Wv bk bo bq bv f1 f2 fb1 fb2 ln1b ln1g ln2b ln2g
  const float* blk[2][16];
  for (int b = 0; b < 2; ++b)
    for (int i = 0; i < 16; ++i) blk[b][i] = (const float*)d_in[10 + b * 16 + i];
  // gat[i]: W a_dst a_src b
  const float* g1[4][4]; const float* g2[4][4];
  for (int i = 0; i < 4; ++i)
    for (int j = 0; j < 4; ++j) {
      g1[i][j] = (const float*)d_in[42 + i * 4 + j];
      g2[i][j] = (const float*)d_in[58 + i * 4 + j];
    }
  const float* lcW = (const float*)d_in[74];
  const float* lcb = (const float*)d_in[75];
  const float* mW1 = (const float*)d_in[76];
  const float* mb1 = (const float*)d_in[77];
  const float* mW2 = (const float*)d_in[78];
  const float* peK = (const float*)d_in[79];
  const float* peQ = (const float*)d_in[80];
  const float* trW[3] = { (const float*)d_in[81], (const float*)d_in[83], (const float*)d_in[85] };
  const float* trB[3] = { (const float*)d_in[82], (const float*)d_in[84], (const float*)d_in[86] };

  // ---- workspace layout (floats), ~112 MB total, with stage-based aliasing
  float* ws   = (float*)d_ws;
  float* XM   = ws + 0;         // 4096x128 ; later reused for mlp out5
  float* XD   = ws + 524288;    // 1024x128
  float* XA   = ws + 655360;    // 3072x128
  float* H1M  = ws + 1048576;   // 4096x512 ; later reused for cat (4096x384)
  float* H1D  = ws + 3145728;   // 1024x512
  float* H1A  = ws + 3670016;   // 3072x512
  float* PS   = ws + 5242880;   // 4096x512 scratch; later Q/K/V/ATT
  float* PD   = ws + 7340032;   // 4096x512 scratch; later O/T1
  float* ASb  = ws + 9437184;   // 4096x4
  float* ADb  = ws + 9453568;
  float* EB   = ws + 9469952;   // 16384x4
  float* WBf  = ws + 9535488;
  float* MB   = ws + 9601024;   // 4096x4
  float* ZB   = ws + 9617408;
  float* X0   = ws + 9633792;   // h2["movie"] 4096x128
  float* X1   = ws + 10158080;
  float* X2   = ws + 10682368;
  float* DEGB = ws + 11206656;  // 4096
  float* LOSS = ws + 11210752;  // 1
  float* AHAT = ws + 11210768;  // 4096x4096

  float* QB  = PS;
  float* KB  = PS + 524288;
  float* VB  = PS + 1048576;
  float* ATT = PS + 1572864;
  float* OB  = PD;
  float* T1  = PD + 524288;
  float* CAT = H1M;
  float* OUT5 = XM;

  float* out      = (float*)d_out;
  float* o_loss   = out;                 // 1
  float* o_logits = out + 1;             // 4096*5
  float* o_xgt    = out + 20481;         // 4096*128
  float* o_peQ    = out + 544769;        // 4096*32
  float* o_peK    = out + 675841;        // 4096*32

  auto gemm = [&](const float* A, const float* B, const float* bias, float* C,
                  int M, int N, int K, int transB, int act, float scale) {
    dim3 g((N + 63) / 64, M / 32);
    gemm_wmma<<<g, 128, 0, stream>>>(A, B, bias, C, M, N, K, transB, act, scale);
  };
  auto fill = [&](float* p, float v, size_t n) {
    fill_kernel<<<(unsigned)((n + 255) / 256), 256, 0, stream>>>(p, v, n);
  };

  // ---- 1. input transforms: x_dict = relu(x @ W + b) ---------------------
  gemm(x_movie,    trW[0], trB[0], XM, NM, HID, IN, 0, 1, 1.f);
  gemm(x_director, trW[1], trB[1], XD, ND, HID, IN, 0, 1, 1.f);
  gemm(x_actor,    trW[2], trB[2], XA, NA, HID, IN, 0, 1, 1.f);

  // ---- 2. GAT layer 1 (4 heads x 128) ------------------------------------
  fill(H1M, 0.f, (size_t)(NM + ND + NA) * H4C);   // H1M/H1D/H1A contiguous
  {
    const float* xs[4] = { XM, XD, XM, XA };
    const float* xd[4] = { XD, XM, XA, XM };
    const int Ns[4] = { NM, ND, NM, NA };
    const int Nd[4] = { ND, NM, NA, NM };
    const int Ec[4] = { 8192, 8192, 16384, 16384 };
    float* outp[4] = { H1D, H1M, H1A, H1M };
    for (int ci = 0; ci < 4; ++ci) {
      const int H = 4, C = 128, E = Ec[ci];
      const int* src = ei[ci], *dst = ei[ci] + E;
      gemm(xs[ci], g1[ci][0], nullptr, PS, Ns[ci], H4C, HID, 0, 0, 1.f);
      gemm(xd[ci], g1[ci][0], nullptr, PD, Nd[ci], H4C, HID, 0, 0, 1.f);
      alpha_red<<<(Ns[ci] * H + 255) / 256, 256, 0, stream>>>(PS, g1[ci][2], ASb, Ns[ci], H, C);
      alpha_red<<<(Nd[ci] * H + 255) / 256, 256, 0, stream>>>(PD, g1[ci][1], ADb, Nd[ci], H, C);
      fill(MB, -3.0e38f, (size_t)Nd[ci] * H);
      fill(ZB, 0.f,      (size_t)Nd[ci] * H);
      edge_max   <<<(E * H + 255) / 256, 256, 0, stream>>>(src, dst, ASb, ADb, EB, MB, E, H);
      edge_expsum<<<(E * H + 255) / 256, 256, 0, stream>>>(dst, EB, MB, ZB, WBf, E, H);
      edge_agg<<<(unsigned)(((size_t)E * H * C + 255) / 256), 256, 0, stream>>>(
          src, dst, WBf, ZB, PS, outp[ci], E, H, C);
      addbias_kernel<<<(Nd[ci] * H4C + 255) / 256, 256, 0, stream>>>(outp[ci], g1[ci][3], Nd[ci], H4C);
    }
  }
  relu_kernel<<<(unsigned)(((size_t)(NM + ND + NA) * H4C + 255) / 256), 256, 0, stream>>>(
      H1M, (size_t)(NM + ND + NA) * H4C);

  // ---- 3. GAT layer 2: only dt == "movie" feeds the transformer ----------
  fill(X0, 0.f, (size_t)NM * HID);
  {
    const int convs[2] = { 1, 3 };                 // (director->movie), (actor->movie)
    const float* xs[2] = { H1D, H1A };
    const int Ns[2] = { ND, NA };
    const int Ec[2] = { 8192, 16384 };
    for (int q = 0; q < 2; ++q) {
      int ci = convs[q];
      const int H = 1, C = 128, E = Ec[q];
      const int* src = ei[ci], *dst = ei[ci] + E;
      gemm(xs[q], g2[ci][0], nullptr, PS, Ns[q], HID, H4C, 0, 0, 1.f);
      gemm(H1M,   g2[ci][0], nullptr, PD, NM,    HID, H4C, 0, 0, 1.f);
      alpha_red<<<(Ns[q] * H + 255) / 256, 256, 0, stream>>>(PS, g2[ci][2], ASb, Ns[q], H, C);
      alpha_red<<<(NM * H + 255) / 256, 256, 0, stream>>>(PD, g2[ci][1], ADb, NM, H, C);
      fill(MB, -3.0e38f, (size_t)NM * H);
      fill(ZB, 0.f,      (size_t)NM * H);
      edge_max   <<<(E * H + 255) / 256, 256, 0, stream>>>(src, dst, ASb, ADb, EB, MB, E, H);
      edge_expsum<<<(E * H + 255) / 256, 256, 0, stream>>>(dst, EB, MB, ZB, WBf, E, H);
      edge_agg<<<(unsigned)(((size_t)E * H * C + 255) / 256), 256, 0, stream>>>(
          src, dst, WBf, ZB, PS, X0, E, H, C);
      addbias_kernel<<<(NM * HID + 255) / 256, 256, 0, stream>>>(X0, g2[ci][3], NM, HID);
    }
  }

  // ---- 4. PE: A_hat = pe_Q @ pe_K^T, loss, deg bias -----------------------
  gemm(peQ, peK, nullptr, AHAT, NM, NM, 32, /*transB=*/1, 0, 1.f);
  fill(LOSS, 0.f, 1);
  pe_loss<<<1024, 256, 0, stream>>>(AHAT, origA, LOSS, NN);
  finalize_loss<<<1, 1, 0, stream>>>(LOSS, alpha_p, o_loss, 1.0f / (4.0f * (float)NN));
  log1p_kernel<<<(NM + 255) / 256, 256, 0, stream>>>(deg, DEGB, NM);

  // ---- 5. graph-transformer blocks ---------------------------------------
  const float ISQ = 0.1767766953f;                 // 1/sqrt(32)
  float* xin = X0;
  float* xouts[2] = { X1, X2 };
  for (int bi = 0; bi < 2; ++bi) {
    const float* const* P = blk[bi];
    gemm(xin, P[2], P[6], QB, NM, HID, HID, 0, 0, 1.f);   // Wq, bq
    gemm(xin, P[0], P[4], KB, NM, HID, HID, 0, 0, 1.f);   // Wk, bk
    gemm(xin, P[3], P[7], VB, NM, HID, HID, 0, 0, 1.f);   // Wv, bv
    flash_attn<<<dim3(NM / 128, 4), 256, 0, stream>>>(QB, KB, VB, AHAT, DEGB, ATT, NM, ISQ, ISQ);
    gemm(ATT, P[1], P[5], OB, NM, HID, HID, 0, 0, 1.f);   // Wo, bo
    ln_residual<<<NM, 128, 0, stream>>>(xin, OB, P[13], P[12], ATT, HID);  // ln1 -> ATT
    gemm(ATT, P[8], P[10], T1, NM, 2 * HID, HID, 0, 1, 1.f);               // relu(f1)
    gemm(T1, P[9], P[11], OB, NM, HID, 2 * HID, 0, 0, 1.f);                // f2
    ln_residual<<<NM, 128, 0, stream>>>(ATT, OB, P[15], P[14], xouts[bi], HID); // ln2
    xin = xouts[bi];
  }

  // ---- 6. head: concat -> lin_cat -> mlp -> softmax -----------------------
  concat3<<<(unsigned)(((size_t)NM * 384 + 255) / 256), 256, 0, stream>>>(X0, X1, X2, CAT, NM);
  gemm(CAT, lcW, lcb, o_xgt, NM, HID, 3 * HID, 0, 1, 1.f);       // x_gt (direct to d_out)
  gemm(o_xgt, mW1, mb1, T1, NM, 2 * HID, HID, 0, 1, 1.f);
  gemm(T1, mW2, nullptr, OUT5, NM, 5, 2 * HID, 0, 0, 1.f);
  softmax5<<<(NM + 255) / 256, 256, 0, stream>>>(OUT5, o_logits, NM);

  // ---- 7. passthrough outputs --------------------------------------------
  hipMemcpyAsync(o_peQ, d_in[80], (size_t)NM * 32 * sizeof(float), hipMemcpyDeviceToDevice, stream);
  hipMemcpyAsync(o_peK, d_in[79], (size_t)NM * 32 * sizeof(float), hipMemcpyDeviceToDevice, stream);
}
